// NHGATModelGAN_29377576304707
// MI455X (gfx1250) — 16-bit and 8-bit matrix operand order repaired and checked by reading, compiled, not run on hardware
//
#include <hip/hip_runtime.h>
#include <hip/hip_bf16.h>
#include <stdint.h>

#define N_NODES 4096
#define F_IN    512
#define NHEADS  4
#define H1      64
#define H2      64
#define H3      32

typedef _Float16 v16h __attribute__((ext_vector_type(16)));
typedef float    v8f  __attribute__((ext_vector_type(8)));
typedef int      v8i  __attribute__((ext_vector_type(8)));
typedef unsigned int u32;
typedef u32 u32x2 __attribute__((ext_vector_type(2)));
typedef u32 u32x4 __attribute__((ext_vector_type(4)));
typedef u32 u32x8 __attribute__((ext_vector_type(8)));

__global__ void k_cvt_f2h(const float* __restrict__ in, _Float16* __restrict__ out, size_t n) {
    size_t i = (size_t)blockIdx.x * blockDim.x + threadIdx.x;
    if (i < n) out[i] = (_Float16)in[i];
}

__global__ void k_cvt_f2u8(const float* __restrict__ in, unsigned char* __restrict__ out, size_t n) {
    size_t i = (size_t)blockIdx.x * blockDim.x + threadIdx.x;
    if (i < n) out[i] = in[i] > 0.f ? 1 : 0;
}

__global__ void k_trcvt(const float* __restrict__ in, _Float16* __restrict__ out, int R, int C) {
    int i = blockIdx.x * blockDim.x + threadIdx.x;
    if (i >= R * C) return;
    int r = i / C, c = i % C;
    out[(size_t)c * R + r] = (_Float16)in[i];
}

#define EPI_RELU    1
#define EPI_SIGMOID 2

__device__ __forceinline__ v16h load_afrag(const _Float16* arow, int k, int hl) {
    const u32* ap = (const u32*)(arow + k) + hl * 8;
    u32x4 alo = *(const u32x4*)ap;
    u32x4 ahi = *(const u32x4*)(ap + 4);
    u32x8 au = { alo[0], alo[1], alo[2], alo[3], ahi[0], ahi[1], ahi[2], ahi[3] };
    return __builtin_bit_cast(v16h, au);
}
__device__ __forceinline__ v16h load_bfrag(const _Float16* brow, int k, int hl) {
    const u32* bp = (const u32*)(brow + k) + hl * 8;
    u32x8 bu = *(const u32x8*)bp;
    return __builtin_bit_cast(v16h, bu);
}
__device__ __forceinline__ v8f wmma_f16(v16h a, v16h b, v8f c) {
    return __builtin_amdgcn_wmma_f32_16x16x32_f16(false, a, false, b, (short)0, c, false, false);
}

__device__ __forceinline__ void epilogue_store(
        float acc, int m, int n, float* C, int ldc, const float* bias,
        const float* res, int ldres, _Float16* Ch, int ldch, int flags) {
    float v = acc + (bias ? bias[n] : 0.f);
    if (res) v += res[(size_t)m * ldres + n];
    if (flags & EPI_RELU)    v = v > 0.f ? v : 0.f;
    if (flags & EPI_SIGMOID) v = 1.f / (1.f + __expf(-v));
    C[(size_t)m * ldc + n] = v;
    if (Ch) Ch[(size_t)m * ldch + n] = (_Float16)v;
}

__global__ void __launch_bounds__(256) k_gemm(
        const _Float16* __restrict__ A, int lda,
        const _Float16* __restrict__ BT, int ldbt,
        float* __restrict__ C, int ldc,
        const float* __restrict__ bias,
        const float* __restrict__ res, int ldres,
        _Float16* __restrict__ Ch, int ldch,
        int Mt, int Nt, int Ncols, int K, int flags) {
    int lane = threadIdx.x & 31;
    int wave = threadIdx.x >> 5;
    int t = blockIdx.x * 8 + wave;
    if (t >= Mt * Nt) return;
    int tm = t / Nt, tn = t % Nt;
    int hl = lane >> 4, l16 = lane & 15;

    int bcol = tn * 16 + l16;
    if (bcol >= Ncols) bcol = Ncols - 1;
    const _Float16* arow = A  + (size_t)(tm * 16 + l16) * lda;
    const _Float16* brow = BT + (size_t)bcol * ldbt;

    v8f acc = {};
    for (int k = 0; k < K; k += 32)
        acc = wmma_f16(load_afrag(arow, k, hl), load_bfrag(brow, k, hl), acc);

    int m0 = tm * 16 + hl * 8;
    int n  = tn * 16 + l16;
    if (n < Ncols) {
        #pragma unroll
        for (int i = 0; i < 8; ++i)
            epilogue_store(acc[i], m0 + i, n, C, ldc, bias, res, ldres, Ch, ldch, flags);
    }
}

__global__ void __launch_bounds__(256) k_gemm4(
        const _Float16* __restrict__ A, int lda,
        const _Float16* __restrict__ BT, int ldbt,
        float* __restrict__ C, int ldc,
        const float* __restrict__ bias,
        const float* __restrict__ res, int ldres,
        _Float16* __restrict__ Ch, int ldch,
        int Mt, int Ng, int K, int flags) {
    int lane = threadIdx.x & 31;
    int wave = threadIdx.x >> 5;
    int t = blockIdx.x * 8 + wave;
    if (t >= Mt * Ng) return;
    int tm = t / Ng, tg = t % Ng;
    int hl = lane >> 4, l16 = lane & 15;

    const _Float16* arow  = A  + (size_t)(tm * 16 + l16) * lda;
    const _Float16* brow0 = BT + (size_t)(tg * 64 + l16) * ldbt;

    v8f acc[4] = {};
    for (int k = 0; k < K; k += 32) {
        v16h av = load_afrag(arow, k, hl);
        #pragma unroll
        for (int j = 0; j < 4; ++j)
            acc[j] = wmma_f16(av, load_bfrag(brow0 + (size_t)j * 16 * ldbt, k, hl), acc[j]);
    }
    int m0 = tm * 16 + hl * 8;
    #pragma unroll
    for (int j = 0; j < 4; ++j) {
        int n = tg * 64 + j * 16 + l16;
        #pragma unroll
        for (int i = 0; i < 8; ++i)
            epilogue_store(acc[j][i], m0 + i, n, C, ldc, bias, res, ldres, Ch, ldch, flags);
    }
}

__global__ void __launch_bounds__(256) k_gemm4_lds(
        const _Float16* __restrict__ A, int lda,
        const _Float16* __restrict__ BT, int ldbt,
        float* __restrict__ C, int ldc,
        const float* __restrict__ bias,
        const float* __restrict__ res, int ldres,
        _Float16* __restrict__ Ch, int ldch,
        int NS, int K, int flags) {
    __shared__ __align__(16) _Float16 sB[2][64 * 64];
    int tid  = threadIdx.x;
    int lane = tid & 31, wave = tid >> 5;
    int bm = blockIdx.x / NS, bs = blockIdx.x % NS;
    int hl = lane >> 4, l16 = lane & 15;

    const _Float16* arow = A + (size_t)(bm * 128 + wave * 16 + l16) * lda;

    int scol = tid >> 2, sseg = tid & 3;
    const _Float16* gsrc = BT + (size_t)(bs * 64 + scol) * ldbt + sseg * 16;
    u32 ldsa0 = (u32)(uintptr_t)&sB[0][scol * 64 + sseg * 16];
    u32 ldsa1 = (u32)(uintptr_t)&sB[1][scol * 64 + sseg * 16];

    {
        unsigned long long ga = (unsigned long long)(uintptr_t)gsrc;
        asm volatile("global_load_async_to_lds_b128 %0, %1, off" :: "v"(ldsa0), "v"(ga) : "memory");
        asm volatile("global_load_async_to_lds_b128 %0, %1, off" :: "v"(ldsa0 + 16u), "v"(ga + 16u) : "memory");
    }

    v8f acc[4] = {};
    int nk = K >> 6;
    for (int kc = 0; kc < nk; ++kc) {
        int buf = kc & 1;
        if (kc + 1 < nk) {
            unsigned long long ga = (unsigned long long)(uintptr_t)(gsrc + (size_t)(kc + 1) * 64);
            u32 ld = buf ? ldsa0 : ldsa1;
            asm volatile("global_load_async_to_lds_b128 %0, %1, off" :: "v"(ld), "v"(ga) : "memory");
            asm volatile("global_load_async_to_lds_b128 %0, %1, off" :: "v"(ld + 16u), "v"(ga + 16u) : "memory");
            asm volatile("s_wait_asynccnt 2" ::: "memory");
        } else {
            asm volatile("s_wait_asynccnt 0" ::: "memory");
        }
        __syncthreads();
        #pragma unroll
        for (int s = 0; s < 2; ++s) {
            v16h av = load_afrag(arow, kc * 64 + s * 32, hl);
            #pragma unroll
            for (int j = 0; j < 4; ++j) {
                const u32* bp = (const u32*)&sB[buf][(j * 16 + l16) * 64 + s * 32 + hl * 16];
                u32x8 bu = *(const u32x8*)bp;
                acc[j] = wmma_f16(av, __builtin_bit_cast(v16h, bu), acc[j]);
            }
        }
        __syncthreads();
    }

    int m0 = bm * 128 + wave * 16 + hl * 8;
    #pragma unroll
    for (int j = 0; j < 4; ++j) {
        int n = bs * 64 + j * 16 + l16;
        #pragma unroll
        for (int i = 0; i < 8; ++i)
            epilogue_store(acc[j][i], m0 + i, n, C, ldc, bias, res, ldres, Ch, ldch, flags);
    }
}

__global__ void __launch_bounds__(256) k_gemm_mask(
        const unsigned char* __restrict__ adj8,
        unsigned char* __restrict__ mask, int n) {
    __shared__ __align__(16) u32 sB[2][1024];
    int tid  = threadIdx.x;
    int lane = tid & 31, wave = tid >> 5;
    int nbc  = n >> 6;
    int bm = blockIdx.x / nbc, bn = blockIdx.x % nbc;
    int hl = lane >> 4, l16 = lane & 15;

    const unsigned char* arow = adj8 + (size_t)(bm * 128 + wave * 16 + l16) * n;

    int scol = tid >> 2, sseg = tid & 3;
    const unsigned char* gsrc = adj8 + (size_t)(bn * 64 + scol) * n + sseg * 16;
    u32 lds0 = (u32)(uintptr_t)&sB[0][scol * 16 + sseg * 4];
    u32 lds1 = (u32)(uintptr_t)&sB[1][scol * 16 + sseg * 4];

    {
        unsigned long long ga = (unsigned long long)(uintptr_t)gsrc;
        asm volatile("global_load_async_to_lds_b128 %0, %1, off" :: "v"(lds0), "v"(ga) : "memory");
    }

    v8i acc[4] = {};
    int nk = n >> 6;
    for (int kc = 0; kc < nk; ++kc) {
        int buf = kc & 1;
        if (kc + 1 < nk) {
            unsigned long long ga = (unsigned long long)(uintptr_t)(gsrc + (size_t)(kc + 1) * 64);
            asm volatile("global_load_async_to_lds_b128 %0, %1, off"
                         :: "v"(buf ? lds0 : lds1), "v"(ga) : "memory");
            asm volatile("s_wait_asynccnt 1" ::: "memory");
        } else {
            asm volatile("s_wait_asynccnt 0" ::: "memory");
        }
        __syncthreads();

        const u32* ap = (const u32*)(arow + kc * 64) + hl * 4;
        u32x4 a0 = *(const u32x4*)ap;
        u32x4 a1 = *(const u32x4*)(ap + 8);
        u32x8 au = { a0[0], a0[1], a0[2], a0[3], a1[0], a1[1], a1[2], a1[3] };
        v8i av = __builtin_bit_cast(v8i, au);

        #pragma unroll
        for (int j = 0; j < 4; ++j) {
            const u32* bp = &sB[buf][(j * 16 + l16) * 16 + hl * 4];
            u32x4 b0 = *(const u32x4*)bp;
            u32x4 b1 = *(const u32x4*)(bp + 8);
            u32x8 bu = { b0[0], b0[1], b0[2], b0[3], b1[0], b1[1], b1[2], b1[3] };
            v8i bv = __builtin_bit_cast(v8i, bu);
            acc[j] = __builtin_amdgcn_wmma_i32_16x16x64_iu8(false, av, false, bv,
                                                            acc[j], false, false);
        }
        __syncthreads();
    }

    int m0 = bm * 128 + wave * 16 + hl * 8;
    #pragma unroll
    for (int j = 0; j < 4; ++j) {
        int col = bn * 64 + j * 16 + l16;
        #pragma unroll
        for (int i = 0; i < 8; ++i) {
            size_t off = (size_t)(m0 + i) * n + col;
            mask[off] = ((acc[j][i] != 0) | (adj8[off] != 0)) ? 1 : 0;
        }
    }
}

__global__ void k_gemv2(const float* __restrict__ Wh, const float* __restrict__ a,
                        float* __restrict__ s1, float* __restrict__ s2, int n, int d) {
    int r = blockIdx.x * blockDim.x + threadIdx.x;
    if (r >= n) return;
    const float* row = Wh + (size_t)r * d;
    float a1 = 0.f, a2 = 0.f;
    for (int c = 0; c < d; ++c) { float v = row[c]; a1 += v * a[c]; a2 += v * a[d + c]; }
    s1[r] = a1; s2[r] = a2;
}

__global__ void k_prow(const float* __restrict__ s1, const float* __restrict__ s2,
                       const unsigned char* __restrict__ mask,
                       _Float16* __restrict__ P, int n) {
    __shared__ float red[256];
    int row = blockIdx.x, tid = threadIdx.x;
    float s1i = s1[row];
    const unsigned char* mrow = mask + (size_t)row * n;
    _Float16* prow = P + (size_t)row * n;

    float mx = -3.0e38f;
    for (int j = tid; j < n; j += 256) {
        if (mrow[j]) {
            float e = s1i + s2[j];
            e = e > 0.f ? e : 0.2f * e;
            mx = fmaxf(mx, e);
        }
    }
    red[tid] = mx; __syncthreads();
    for (int s = 128; s > 0; s >>= 1) { if (tid < s) red[tid] = fmaxf(red[tid], red[tid + s]); __syncthreads(); }
    float rm = red[0]; __syncthreads();

    float sum = 0.f;
    for (int j = tid; j < n; j += 256) {
        float p = 0.f;
        if (mrow[j]) {
            float e = s1i + s2[j];
            e = e > 0.f ? e : 0.2f * e;
            p = __expf(e - rm);
        }
        prow[j] = (_Float16)p;
        sum += p;
    }
    red[tid] = sum; __syncthreads();
    for (int s = 128; s > 0; s >>= 1) { if (tid < s) red[tid] += red[tid + s]; __syncthreads(); }
    float inv = red[0] > 0.f ? 1.f / red[0] : 0.f;
    for (int j = tid; j < n; j += 256) prow[j] = (_Float16)((float)prow[j] * inv);
}

__global__ void k_z(const float* __restrict__ mu, const float* __restrict__ lv,
                    const float* __restrict__ eps, float* __restrict__ z,
                    _Float16* __restrict__ zh, int n) {
    int i = blockIdx.x * blockDim.x + threadIdx.x;
    if (i >= n) return;
    float v = eps[i] * __expf(lv[i]) + mu[i];
    z[i] = v; zh[i] = (_Float16)v;
}

__global__ void k_disc_out(const float* __restrict__ h2, const float* __restrict__ w,
                           const float* __restrict__ b, float* __restrict__ out, int n) {
    int r = blockIdx.x * blockDim.x + threadIdx.x;
    if (r >= n) return;
    const float* row = h2 + (size_t)r * 32;
    float acc = b[0];
    for (int c = 0; c < 32; ++c) acc += row[c] * w[c];
    out[r] = 1.f / (1.f + __expf(-acc));
}

extern "C" void kernel_launch(void* const* d_in, const int* in_sizes, int n_in,
                              void* d_out, int out_size, void* d_ws, size_t ws_size,
                              hipStream_t stream) {
    const int N = N_NODES;
    const float* x       = (const float*)d_in[0];
    const float* adj     = (const float*)d_in[1];
    const float* W_heads = (const float*)d_in[2];
    const float* a_heads = (const float*)d_in[3];
    const float* W_att   = (const float*)d_in[4];
    const float* a_att   = (const float*)d_in[5];
    const float* gc1_W   = (const float*)d_in[6];
    const float* gc1_fcW = (const float*)d_in[7];
    const float* gc1_fcb = (const float*)d_in[8];
    const float* gc2_W   = (const float*)d_in[9];
    const float* gc2_fcW = (const float*)d_in[10];
    const float* gc2_fcb = (const float*)d_in[11];
    const float* gc2_resW= (const float*)d_in[12];
    const float* gc2_resb= (const float*)d_in[13];
    const float* gc3_W   = (const float*)d_in[14];
    const float* gc3_fcW = (const float*)d_in[15];
    const float* gc3_fcb = (const float*)d_in[16];
    const float* gc3_resW= (const float*)d_in[17];
    const float* gc3_resb= (const float*)d_in[18];
    const float* dW1 = (const float*)d_in[19];
    const float* db1 = (const float*)d_in[20];
    const float* dW2 = (const float*)d_in[21];
    const float* db2 = (const float*)d_in[22];
    const float* dW3 = (const float*)d_in[23];
    const float* db3 = (const float*)d_in[24];
    const float* eps    = (const float*)d_in[25];
    const float* z_real = (const float*)d_in[26];

    float* out      = (float*)d_out;
    float* pred     = out;
    float* dis_real = out + (size_t)N * N;
    float* dis_fake = dis_real + N;
    float* mu       = dis_fake + N;
    float* logvar   = mu + (size_t)N * H3;

    char* wp = (char*)d_ws;
    auto alloc = [&](size_t bytes) -> void* {
        void* r = (void*)wp;
        wp += (bytes + 255) & ~(size_t)255;
        return r;
    };
    _Float16* adj_h  = (_Float16*)alloc((size_t)N * N * 2);
    _Float16* P      = (_Float16*)alloc((size_t)N * N * 2);
    unsigned char* adj8  = (unsigned char*)alloc((size_t)N * N);
    unsigned char* maskb = (unsigned char*)alloc((size_t)N * N);
    _Float16* xh     = (_Float16*)alloc((size_t)N * F_IN * 2);
    _Float16* Wt     = (_Float16*)alloc(65536 * 2);
    float*    Wh     = (float*)alloc((size_t)N * 64 * 4);
    _Float16* WhT    = (_Float16*)alloc((size_t)64 * N * 2);
    float*    s1     = (float*)alloc(N * 4);
    float*    s2     = (float*)alloc(N * 4);
    float*    hcat   = (float*)alloc((size_t)N * 256 * 4);
    _Float16* hcat_h = (_Float16*)alloc((size_t)N * 256 * 2);
    float*    gatt   = (float*)alloc((size_t)N * 64 * 4);
    _Float16* gatt_h = (_Float16*)alloc((size_t)N * 64 * 2);
    float*    XWc    = (float*)alloc((size_t)N * 192 * 4);
    _Float16* XWT    = (_Float16*)alloc((size_t)192 * N * 2);
    float*    feats  = (float*)alloc((size_t)N * 192 * 4);
    _Float16* feats_h= (_Float16*)alloc((size_t)N * 192 * 2);
    float*    h1     = (float*)alloc((size_t)N * 64 * 4);
    _Float16* h1_h   = (_Float16*)alloc((size_t)N * 64 * 2);
    float*    resbuf = (float*)alloc((size_t)N * 32 * 4);
    float*    zbuf   = (float*)alloc((size_t)N * 32 * 4);
    _Float16* zh     = (_Float16*)alloc((size_t)N * 32 * 2);
    _Float16* zrl_h  = (_Float16*)alloc((size_t)N * 32 * 2);
    float*    hb1    = (float*)alloc((size_t)N * 32 * 4);
    _Float16* hb1_h  = (_Float16*)alloc((size_t)N * 32 * 2);
    float*    hb2    = (float*)alloc((size_t)N * 32 * 4);

    auto cvt = [&](const float* in, _Float16* o, size_t n) {
        k_cvt_f2h<<<dim3((unsigned)((n + 255) / 256)), dim3(256), 0, stream>>>(in, o, n);
    };
    auto trcvt = [&](const float* in, _Float16* o, int R, int C) {
        int n = R * C;
        k_trcvt<<<dim3((n + 255) / 256), dim3(256), 0, stream>>>(in, o, R, C);
    };
    auto gemm = [&](const _Float16* A, int lda, const _Float16* BT, int ldbt,
                    float* C, int ldc, const float* bias,
                    const float* res, int ldres, _Float16* Ch, int ldch,
                    int M, int Ncols, int K, int flags) {
        int Mt = M / 16;
        if ((Ncols & 63) == 0 && (M & 127) == 0 && (K & 63) == 0 && K >= 128) {
            int NS = Ncols / 64;
            k_gemm4_lds<<<dim3((M / 128) * NS), dim3(256), 0, stream>>>(
                A, lda, BT, ldbt, C, ldc, bias, res, ldres, Ch, ldch, NS, K, flags);
        } else if ((Ncols & 63) == 0) {
            int Ng = Ncols / 64;
            k_gemm4<<<dim3((Mt * Ng + 7) / 8), dim3(256), 0, stream>>>(
                A, lda, BT, ldbt, C, ldc, bias, res, ldres, Ch, ldch, Mt, Ng, K, flags);
        } else {
            int Nt = (Ncols + 15) / 16;
            k_gemm<<<dim3((Mt * Nt + 7) / 8), dim3(256), 0, stream>>>(
                A, lda, BT, ldbt, C, ldc, bias, res, ldres, Ch, ldch, Mt, Nt, Ncols, K, flags);
        }
    };

    k_cvt_f2u8<<<dim3((unsigned)(((size_t)N * N + 255) / 256)), dim3(256), 0, stream>>>(
        adj, adj8, (size_t)N * N);
    cvt(adj, adj_h, (size_t)N * N);
    cvt(x, xh, (size_t)N * F_IN);

    k_gemm_mask<<<dim3((N / 128) * (N / 64)), dim3(256), 0, stream>>>(adj8, maskb, N);

    for (int h = 0; h < NHEADS; ++h) {
        trcvt(W_heads + (size_t)h * F_IN * H1, Wt, F_IN, H1);
        gemm(xh, F_IN, Wt, F_IN, Wh, H1, nullptr, nullptr, 0, nullptr, 0, N, H1, F_IN, 0);
        trcvt(Wh, WhT, N, H1);
        k_gemv2<<<dim3(N / 256), dim3(256), 0, stream>>>(Wh, a_heads + h * 2 * H1, s1, s2, N, H1);
        k_prow<<<dim3(N), dim3(256), 0, stream>>>(s1, s2, maskb, P, N);
        gemm(P, N, WhT, N, hcat + h * H1, 256, nullptr, nullptr, 0,
             hcat_h + h * H1, 256, N, H1, N, EPI_RELU);
    }

    trcvt(W_att, Wt, NHEADS * H1, H1);
    gemm(hcat_h, 256, Wt, 256, Wh, H1, nullptr, nullptr, 0, nullptr, 0, N, H1, NHEADS * H1, 0);
    trcvt(Wh, WhT, N, H1);
    k_gemv2<<<dim3(N / 256), dim3(256), 0, stream>>>(Wh, a_att, s1, s2, N, H1);
    k_prow<<<dim3(N), dim3(256), 0, stream>>>(s1, s2, maskb, P, N);
    gemm(P, N, WhT, N, gatt, H1, nullptr, nullptr, 0, gatt_h, H1, N, H1, N, EPI_RELU);

    for (int i = 0; i < 3; ++i) {
        trcvt(gc1_W + (size_t)i * H1 * H2, Wt, H1, H2);
        gemm(gatt_h, H1, Wt, H1, XWc + i * H2, 192, nullptr, nullptr, 0, nullptr, 0,
             N, H2, H1, 0);
    }
    trcvt(XWc, XWT, N, 192);
    gemm(adj_h, N, XWT, N, feats, 192, nullptr, nullptr, 0, feats_h, 192, N, 192, N, EPI_RELU);
    trcvt(gc1_fcW, Wt, 3 * H2, H2);
    gemm(feats_h, 192, Wt, 192, h1, H2, gc1_fcb, gatt, H2, h1_h, H2, N, H2, 3 * H2, 0);

    const float* gcW[2] = { gc2_W, gc3_W };
    const float* fcW[2] = { gc2_fcW, gc3_fcW };
    const float* fcb[2] = { gc2_fcb, gc3_fcb };
    const float* rW[2]  = { gc2_resW, gc3_resW };
    const float* rb[2]  = { gc2_resb, gc3_resb };
    float* dst[2] = { mu, logvar };
    for (int g = 0; g < 2; ++g)
        for (int i = 0; i < 3; ++i) {
            trcvt(gcW[g] + (size_t)i * H2 * H3, Wt, H2, H3);
            gemm(h1_h, H2, Wt, H2, XWc + g * 96 + i * H3, 192, nullptr, nullptr, 0,
                 nullptr, 0, N, H3, H2, 0);
        }
    trcvt(XWc, XWT, N, 192);
    gemm(adj_h, N, XWT, N, feats, 192, nullptr, nullptr, 0, feats_h, 192, N, 192, N, EPI_RELU);
    for (int g = 0; g < 2; ++g) {
        trcvt(rW[g], Wt, H2, H3);
        gemm(h1_h, H2, Wt, H2, resbuf, H3, rb[g], nullptr, 0, nullptr, 0, N, H3, H2, 0);
        trcvt(fcW[g], Wt, 3 * H3, H3);
        gemm(feats_h + g * 96, 192, Wt, 3 * H3, dst[g], H3, fcb[g], resbuf, H3, nullptr, 0,
             N, H3, 3 * H3, 0);
    }

    k_z<<<dim3(N * H3 / 256), dim3(256), 0, stream>>>(mu, logvar, eps, zbuf, zh, N * H3);

    gemm(zh, H3, zh, H3, pred, N, nullptr, nullptr, 0, nullptr, 0, N, N, H3, EPI_SIGMOID);

    cvt(z_real, zrl_h, (size_t)N * H3);
    const _Float16* din[2] = { zrl_h, zh };
    float* dout[2] = { dis_real, dis_fake };
    for (int g = 0; g < 2; ++g) {
        trcvt(dW1, Wt, H3, H3);
        gemm(din[g], H3, Wt, H3, hb1, H3, db1, nullptr, 0, hb1_h, H3, N, H3, H3, EPI_RELU);
        trcvt(dW2, Wt, H3, H3);
        gemm(hb1_h, H3, Wt, H3, hb2, H3, db2, nullptr, 0, nullptr, 0, N, H3, H3, EPI_RELU);
        k_disc_out<<<dim3(N / 256), dim3(256), 0, stream>>>(hb2, dW3, db3, dout[g], N);
    }
    (void)in_sizes; (void)n_in; (void)out_size; (void)ws_size;
}
